// VitPoseBackboneMoeMLP_8581344657780
// MI455X (gfx1250) — compile-verified, matrix operand order checked by reading, not run on hardware
//
#include <hip/hip_runtime.h>
#include <hip/hip_bf16.h>

// Problem constants (match reference)
#define HIDDEN   768
#define HID_F    3072
#define SHARED_N 576
#define PART     192
#define SSEQ     1024
#define NB       32        // batch
#define TM       32        // tokens per workgroup (two 16-row WMMA tiles)

typedef __attribute__((ext_vector_type(16))) __bf16 v16bf;
typedef __attribute__((ext_vector_type(8)))  __bf16 v8bf;
typedef __attribute__((ext_vector_type(8)))  float  v8f;

union Frag { v16bf v; v8bf h[2]; };

// LDS strides (elements); stride*2 bytes multiple of 16 for aligned b128
#define LDX 776    // 768 + 8
#define LDH 3080   // 3072 + 8

// Swizzled-weight geometry: fragment = 16(N) x 32(K) bf16, stored as
// 32 lanes x 16 contiguous bf16 (512 elements = 1024 B per fragment).
// Fragment order: n_tile-major, k_tile inner  -> streaming reads over K.
#define FRAG_ELEMS 512

// ---------------- Pre-pass: fp32 row-major [K][N] -> swizzled bf16 fragments
__global__ __launch_bounds__(256)
void swizzle_weights_kernel(const float* __restrict__ src, __bf16* __restrict__ dst,
                            int n_tiles, int k_tiles, int src_ld,
                            size_t src_batch_stride, size_t dst_batch_stride)
{
    src += (size_t)blockIdx.y * src_batch_stride;
    dst += (size_t)blockIdx.y * dst_batch_stride;

    const int gt   = blockIdx.x * 256 + threadIdx.x;   // (frag, lane)
    const int frag = gt >> 5;
    const int lane = gt & 31;
    if (frag >= n_tiles * k_tiles) return;

    const int ktile = frag % k_tiles;
    const int ntile = frag / k_tiles;
    const int n     = ntile * 16 + (lane & 15);
    const int k0    = ktile * 32 + ((lane >> 4) & 1) * 8;   // lane-half K offset

    Frag f;
    #pragma unroll
    for (int j = 0; j < 8; ++j) {
        f.v[j]     = (__bf16)src[(size_t)(k0 + j)      * src_ld + n];
        f.v[8 + j] = (__bf16)src[(size_t)(k0 + 16 + j) * src_ld + n];
    }
    __bf16* d = dst + (size_t)frag * FRAG_ELEMS + lane * 16;
    reinterpret_cast<v8bf*>(d)[0] = f.h[0];
    reinterpret_cast<v8bf*>(d)[1] = f.h[1];
}

// ---------------- Fused MoE MLP, bf16 WMMA, h kept in LDS ----------------
__global__ __launch_bounds__(256)
void moe_mlp_wmma_kernel(const float* __restrict__ X,
                         const long long* __restrict__ idx,
                         const __bf16* __restrict__ W1s,   // 192 n_tiles x 24 k_tiles
                         const __bf16* __restrict__ W2s,   //  36 n_tiles x 96 k_tiles
                         const __bf16* __restrict__ Wes,   // 6 x (12 n_tiles x 96 k_tiles)
                         const float* __restrict__ b1,
                         const float* __restrict__ b2,
                         const float* __restrict__ be,
                         float* __restrict__ out)
{
    __shared__ __align__(16) __bf16 lds_X[TM * LDX];   //  49,664 B
    __shared__ __align__(16) __bf16 lds_H[TM * LDH];   // 197,120 B  (CDNA5 320KB LDS)

    const int tid  = threadIdx.x;
    const int lane = tid & 31;
    const int wv   = tid >> 5;            // wave 0..7
    const int half = (lane >> 4) & 1;
    const int lm   = lane & 15;
    const int koff = half * 8;

    const int blk = blockIdx.x;           // 0..1023
    const int b   = blk >> 5;             // 32 token-tiles per batch
    const int s0  = (blk & 31) * TM;
    const int e   = (int)idx[b];

    // ---- Stage X tile [32 x 768] fp32 -> bf16 ----
    {
        const int r  = tid >> 3;          // 0..31
        const int c0 = (tid & 7) * 96;    // 8 threads/row * 96 cols
        const float* src = X + ((size_t)b * SSEQ + (s0 + r)) * HIDDEN + c0;
        #pragma unroll
        for (int i = 0; i < 24; ++i) {
            float4 v = reinterpret_cast<const float4*>(src)[i];
            __bf16* d = &lds_X[r * LDX + c0 + i * 4];
            d[0] = (__bf16)v.x; d[1] = (__bf16)v.y;
            d[2] = (__bf16)v.z; d[3] = (__bf16)v.w;
        }
    }
    __syncthreads();

    // ===== Phase 1: h = gelu(X @ W1 + b1) -> lds_H (bf16) =====
    // Each wave: 2 N-tiles (n0, n0+128) x 2 M-tiles. 12 nb-blocks of 256 cols.
    for (int nb = 0; nb < HID_F; nb += 256) {
        const int n0 = nb + wv * 16;
        v8f acc[4] = {v8f{}, v8f{}, v8f{}, v8f{}};   // [n-sub*2 + m-tile]
        const __bf16* wp0 = W1s + ((size_t)(n0 >> 4) * 24) * FRAG_ELEMS + lane * 16;
        const __bf16* wp1 = W1s + ((size_t)((n0 + 128) >> 4) * 24) * FRAG_ELEMS + lane * 16;

        for (int kt = 0; kt < 24; ++kt) {
            Frag a0, a1, bm0, bm1;
            const __bf16* ap = &lds_X[lm * LDX + kt * 32 + koff];
            a0.h[0] = *reinterpret_cast<const v8bf*>(ap);
            a0.h[1] = *reinterpret_cast<const v8bf*>(ap + 16);
            a1.h[0] = *reinterpret_cast<const v8bf*>(ap + 16 * LDX);
            a1.h[1] = *reinterpret_cast<const v8bf*>(ap + 16 * LDX + 16);
            const v8bf* bp0 = reinterpret_cast<const v8bf*>(wp0 + (size_t)kt * FRAG_ELEMS);
            bm0.h[0] = bp0[0]; bm0.h[1] = bp0[1];
            const v8bf* bp1 = reinterpret_cast<const v8bf*>(wp1 + (size_t)kt * FRAG_ELEMS);
            bm1.h[0] = bp1[0]; bm1.h[1] = bp1[1];

            acc[0] = __builtin_amdgcn_wmma_f32_16x16x32_bf16(false, a0.v, false, bm0.v, (short)0, acc[0], false, false);
            acc[1] = __builtin_amdgcn_wmma_f32_16x16x32_bf16(false, a1.v, false, bm0.v, (short)0, acc[1], false, false);
            acc[2] = __builtin_amdgcn_wmma_f32_16x16x32_bf16(false, a0.v, false, bm1.v, (short)0, acc[2], false, false);
            acc[3] = __builtin_amdgcn_wmma_f32_16x16x32_bf16(false, a1.v, false, bm1.v, (short)0, acc[3], false, false);
        }
        // bias + exact GELU -> lds_H
        #pragma unroll
        for (int ns = 0; ns < 2; ++ns) {
            const int n = n0 + ns * 128 + lm;
            const float bias = b1[n];
            #pragma unroll
            for (int mt = 0; mt < 2; ++mt) {
                #pragma unroll
                for (int r = 0; r < 8; ++r) {
                    const int m = mt * 16 + r + half * 8;
                    float x = acc[ns * 2 + mt][r] + bias;
                    float g = 0.5f * x * (1.0f + erff(x * 0.70710678118654752f));
                    lds_H[m * LDH + n] = (__bf16)g;
                }
            }
        }
    }
    __syncthreads();

    // ===== Phase 2: out = [ h@W2 + b2 , h@We[e] + be[e] ] =====
    const __bf16* WesE = Wes + (size_t)e * (12 * 96) * FRAG_ELEMS;
    for (int nb = 0; nb < HIDDEN; nb += 256) {     // 3 blocks
        const int n0a = nb + wv * 16;
        const int n0b = n0a + 128;
        v8f acc[4] = {v8f{}, v8f{}, v8f{}, v8f{}};
        const __bf16* wpa = (n0a < SHARED_N)
            ? W2s  + ((size_t)(n0a >> 4) * 96) * FRAG_ELEMS + lane * 16
            : WesE + ((size_t)((n0a - SHARED_N) >> 4) * 96) * FRAG_ELEMS + lane * 16;
        const __bf16* wpb = (n0b < SHARED_N)
            ? W2s  + ((size_t)(n0b >> 4) * 96) * FRAG_ELEMS + lane * 16
            : WesE + ((size_t)((n0b - SHARED_N) >> 4) * 96) * FRAG_ELEMS + lane * 16;

        for (int kt = 0; kt < 96; ++kt) {
            Frag a0, a1, bm0, bm1;
            const __bf16* ap = &lds_H[lm * LDH + kt * 32 + koff];
            a0.h[0] = *reinterpret_cast<const v8bf*>(ap);
            a0.h[1] = *reinterpret_cast<const v8bf*>(ap + 16);
            a1.h[0] = *reinterpret_cast<const v8bf*>(ap + 16 * LDH);
            a1.h[1] = *reinterpret_cast<const v8bf*>(ap + 16 * LDH + 16);
            const v8bf* bp0 = reinterpret_cast<const v8bf*>(wpa + (size_t)kt * FRAG_ELEMS);
            bm0.h[0] = bp0[0]; bm0.h[1] = bp0[1];
            const v8bf* bp1 = reinterpret_cast<const v8bf*>(wpb + (size_t)kt * FRAG_ELEMS);
            bm1.h[0] = bp1[0]; bm1.h[1] = bp1[1];

            acc[0] = __builtin_amdgcn_wmma_f32_16x16x32_bf16(false, a0.v, false, bm0.v, (short)0, acc[0], false, false);
            acc[1] = __builtin_amdgcn_wmma_f32_16x16x32_bf16(false, a1.v, false, bm0.v, (short)0, acc[1], false, false);
            acc[2] = __builtin_amdgcn_wmma_f32_16x16x32_bf16(false, a0.v, false, bm1.v, (short)0, acc[2], false, false);
            acc[3] = __builtin_amdgcn_wmma_f32_16x16x32_bf16(false, a1.v, false, bm1.v, (short)0, acc[3], false, false);
        }
        // bias + fp32 store (concat layout == our column ordering)
        #pragma unroll
        for (int ns = 0; ns < 2; ++ns) {
            const int n = (ns ? n0b : n0a) + lm;
            const float bias = (n < SHARED_N) ? b2[n] : be[e * PART + (n - SHARED_N)];
            #pragma unroll
            for (int mt = 0; mt < 2; ++mt) {
                #pragma unroll
                for (int r = 0; r < 8; ++r) {
                    const int m = mt * 16 + r + half * 8;
                    out[((size_t)b * SSEQ + (s0 + m)) * HIDDEN + n] = acc[ns * 2 + mt][r] + bias;
                }
            }
        }
    }
}

extern "C" void kernel_launch(void* const* d_in, const int* in_sizes, int n_in,
                              void* d_out, int out_size, void* d_ws, size_t ws_size,
                              hipStream_t stream) {
    (void)in_sizes; (void)n_in; (void)out_size; (void)ws_size;
    const float*     X   = (const float*)d_in[0];
    const long long* idx = (const long long*)d_in[1];
    const float*     W1  = (const float*)d_in[2];
    const float*     b1  = (const float*)d_in[3];
    const float*     W2  = (const float*)d_in[4];
    const float*     b2  = (const float*)d_in[5];
    const float*     We  = (const float*)d_in[6];
    const float*     be  = (const float*)d_in[7];
    float* out = (float*)d_out;

    // Workspace layout (bf16 elements): ~15.3 MB total (fits easily in L2)
    __bf16* W1s = (__bf16*)d_ws;                               // 192*24*512
    __bf16* W2s = W1s + (size_t)192 * 24 * FRAG_ELEMS;         //  36*96*512
    __bf16* Wes = W2s + (size_t)36  * 96 * FRAG_ELEMS;         // 6*12*96*512

    // Pre-swizzle weights (runs every call; deterministic)
    swizzle_weights_kernel<<<dim3((192 * 24 * 32) / 256, 1), 256, 0, stream>>>(
        W1, W1s, 192, 24, HID_F, 0, 0);
    swizzle_weights_kernel<<<dim3((36 * 96 * 32) / 256, 1), 256, 0, stream>>>(
        W2, W2s, 36, 96, SHARED_N, 0, 0);
    swizzle_weights_kernel<<<dim3((12 * 96 * 32) / 256, 6), 256, 0, stream>>>(
        We, Wes, 12, 96, PART, (size_t)HID_F * PART, (size_t)12 * 96 * FRAG_ELEMS);

    // Fused MoE MLP
    moe_mlp_wmma_kernel<<<NB * (SSEQ / TM), 256, 0, stream>>>(
        X, idx, W1s, W2s, Wes, b1, b2, be, out);
}
